// MultiHeadDotProductAttention_76725295776285
// MI455X (gfx1250) — matrix operand order repaired and checked by reading, not compiled, not run on hardware
//
#include <hip/hip_runtime.h>

#define BQ  4096   // sequence length (Q == K)
#define NB  2      // batch
#define NF  512    // features
#define NH  8      // heads
#define ND  64     // head dim
#define BL  (NB*BQ)   // 8192 flattened rows
#define NHD (NH*ND)   // 512

typedef __attribute__((ext_vector_type(16))) __bf16 v16bf;
typedef __attribute__((ext_vector_type(8)))  __bf16 v8bf;
typedef __attribute__((ext_vector_type(8)))  float  v8f;

static __device__ __forceinline__ v8f wmma_bf16(v16bf a, v16bf b, v8f c) {
  // D = A(16x32 bf16) * B(32x16 bf16) + C(16x16 f32)
  return __builtin_amdgcn_wmma_f32_16x16x32_bf16(false, a, false, b, (short)0, c,
                                                 false, false);
}

static __device__ __forceinline__ v16bf load_b_frag(const __bf16* p) {
  // two 8-bf16 chunks per lane at p and p+16 -> two 16-byte loads
  v8bf c0 = *(const v8bf*)(p);
  v8bf c1 = *(const v8bf*)(p + 16);
  v16bf r;
#pragma unroll
  for (int i = 0; i < 8; i++) { r[i] = c0[i]; r[8 + i] = c1[i]; }
  return r;
}

// ---------------------------------------------------------------------------
// Kernel 0: weight transpose + f32->bf16.  Wt[n*512 + k] = (bf16)W[k*512 + n]
// (512x512, done once; writes are fully coalesced)
// ---------------------------------------------------------------------------
__global__ __launch_bounds__(256)
void convert_w_kernel(const float* __restrict__ W, __bf16* __restrict__ Wt)
{
  const int idx = blockIdx.x * 256 + threadIdx.x;   // 0 .. 512*512-1
  const int n = idx >> 9;
  const int k = idx & 511;
  Wt[(size_t)n * 512 + k] = (__bf16)W[(size_t)k * 512 + n];
}

// ---------------------------------------------------------------------------
// Kernel 1: QKV projection  [BL, F] x Wt[N,K] -> bf16
//   vtrans==0: Out[B][H][L][D]   (q, k)
//   vtrans==1: Out[B][H][D][L]   (v, pre-transposed for the P.V WMMA)
// 4 waves / block; each wave computes a 16(M) x 64(N) tile.
// ---------------------------------------------------------------------------
__global__ __launch_bounds__(128)
void qkv_proj_kernel(const float* __restrict__ X,    // [BL, NF] f32
                     const __bf16* __restrict__ Wt,  // [NHD(n), NF(k)] bf16
                     __bf16* __restrict__ Out,
                     int vtrans)
{
  const int lane = threadIdx.x & 31;
  const int wave = threadIdx.x >> 5;
  const int m  = lane & 15;
  const int hi = lane >> 4;
  const int row0 = blockIdx.x * 64 + wave * 16;
  const int col0 = blockIdx.y * 64;

  v8f c[4] = {};

  for (int kk = 0; kk < NF; kk += 32) {
    // A fragment (16x32): lane holds row m, k = hi*8+i (i<8) / 16+hi*8+(i-8)
    const float* xr = X + (size_t)(row0 + m) * NF + kk;
    float4 x0 = *(const float4*)(xr + hi * 8);
    float4 x1 = *(const float4*)(xr + hi * 8 + 4);
    float4 x2 = *(const float4*)(xr + 16 + hi * 8);
    float4 x3 = *(const float4*)(xr + 16 + hi * 8 + 4);
    v16bf a;
    a[0]=(__bf16)x0.x;  a[1]=(__bf16)x0.y;  a[2]=(__bf16)x0.z;  a[3]=(__bf16)x0.w;
    a[4]=(__bf16)x1.x;  a[5]=(__bf16)x1.y;  a[6]=(__bf16)x1.z;  a[7]=(__bf16)x1.w;
    a[8]=(__bf16)x2.x;  a[9]=(__bf16)x2.y;  a[10]=(__bf16)x2.z; a[11]=(__bf16)x2.w;
    a[12]=(__bf16)x3.x; a[13]=(__bf16)x3.y; a[14]=(__bf16)x3.z; a[15]=(__bf16)x3.w;

#pragma unroll
    for (int ns = 0; ns < 4; ns++) {
      // B fragment: lane holds col n, contraction k = kk + hi*8+i (i<8) / 16+hi*8+(i-8)
      const int n = col0 + ns * 16 + m;
      v16bf bb = load_b_frag(Wt + (size_t)n * NF + kk + hi * 8);
      c[ns] = wmma_bf16(a, bb, c[ns]);
    }
  }

  // C layout: lane L, vgpr r -> row (L/16)*8 + r, col L%16
  if (!vtrans) {
#pragma unroll
    for (int ns = 0; ns < 4; ns++) {
#pragma unroll
      for (int r = 0; r < 8; r++) {
        const int grow = row0 + hi * 8 + r;      // b*BQ + l
        const int gcol = col0 + ns * 16 + m;     // h*64 + d
        const int b_ = grow >> 12;
        const int l  = grow & 4095;
        const int h  = gcol >> 6;
        const int d  = gcol & 63;
        Out[(((size_t)b_ * NH + h) * BQ + l) * ND + d] = (__bf16)c[ns][r];
      }
    }
  } else {
    // transposed store: 8 consecutive l's per lane -> one packed 16B store
    const int b_ = row0 >> 12;
    const int l0 = (row0 & 4095) + hi * 8;
#pragma unroll
    for (int ns = 0; ns < 4; ns++) {
      const int gcol = col0 + ns * 16 + m;       // h*64 + d
      const int h = gcol >> 6;
      const int d = gcol & 63;
      v8bf pack;
#pragma unroll
      for (int r = 0; r < 8; r++) pack[r] = (__bf16)c[ns][r];
      *(v8bf*)&Out[(((size_t)b_ * NH + h) * ND + d) * BQ + l0] = pack;
    }
  }
}

// ---------------------------------------------------------------------------
// Kernel 2: flash attention. grid = (Q/64, H, B), 4 waves/block.
// One wave owns a 16-query tile; iterates 32-key blocks with online softmax.
// Q,K in [B][H][L][D]; V in [B][H][D][L] (pre-transposed).
// ---------------------------------------------------------------------------
__global__ __launch_bounds__(128)
void flash_attn_kernel(const __bf16* __restrict__ Qb,  // [B][H][L][D]
                       const __bf16* __restrict__ Kb,  // [B][H][L][D]
                       const __bf16* __restrict__ Vt,  // [B][H][D][L]
                       __bf16* __restrict__ Attn)      // [B*Q, NHD]
{
  __shared__ __bf16 psh[4][16 * 32];   // per-wave P staging tile
  const int lane = threadIdx.x & 31;
  const int wave = threadIdx.x >> 5;
  const int m  = lane & 15;
  const int hi = lane >> 4;
  const int h  = blockIdx.y;
  const int b  = blockIdx.z;
  const int q0 = blockIdx.x * 64 + wave * 16;

  const __bf16* Qp = Qb + (((size_t)b * NH + h) * BQ) * ND;
  const __bf16* Kp = Kb + (((size_t)b * NH + h) * BQ) * ND;
  const __bf16* Vp = Vt + (((size_t)b * NH + h) * ND) * BQ;

  // Q A-fragments for two 32-wide D chunks (reused across all key blocks)
  v16bf aq[2];
#pragma unroll
  for (int dc = 0; dc < 2; dc++) {
    const __bf16* qr = Qp + (size_t)(q0 + m) * ND + dc * 32;
    v8bf c0 = *(const v8bf*)(qr + hi * 8);
    v8bf c1 = *(const v8bf*)(qr + 16 + hi * 8);
#pragma unroll
    for (int i = 0; i < 8; i++) { aq[dc][i] = c0[i]; aq[dc][8 + i] = c1[i]; }
  }

  float mrow[8], lrow[8];
#pragma unroll
  for (int r = 0; r < 8; r++) { mrow[r] = -3.0e38f; lrow[r] = 0.0f; }
  v8f o[4] = {};

  for (int kb = 0; kb < BQ; kb += 32) {
    // prefetch next key block (K rows; V transposed columns) into cache
    if (kb + 32 < BQ) {
      __builtin_prefetch(Kp + (size_t)(kb + 32 + lane) * ND, 0, 3);
      __builtin_prefetch(Vp + (size_t)lane * BQ + kb + 32, 0, 3);
      __builtin_prefetch(Vp + (size_t)(lane + 32) * BQ + kb + 32, 0, 3);
    }

    // ---- S = Q . K^T for this 16x32 block (two 16-key halves) ----
    v8f s[2];
#pragma unroll
    for (int kh = 0; kh < 2; kh++) {
      v8f acc = {};
#pragma unroll
      for (int dc = 0; dc < 2; dc++) {
        // B fragment: lane holds key n = kb+kh*16+m, contraction d = dc*32+hi*8+i (i<8) / dc*32+16+hi*8+(i-8)
        v16bf bk = load_b_frag(Kp + (size_t)(kb + kh * 16 + m) * ND + dc * 32 + hi * 8);
        acc = wmma_bf16(aq[dc], bk, acc);
      }
      s[kh] = acc;
    }

    // ---- online softmax (row = hi*8 + r lives across one 16-lane half) ----
    float scale[8];
#pragma unroll
    for (int r = 0; r < 8; r++) {
      float mx = fmaxf(s[0][r], s[1][r]);
      mx = fmaxf(mx, __shfl_xor(mx, 1, 16));
      mx = fmaxf(mx, __shfl_xor(mx, 2, 16));
      mx = fmaxf(mx, __shfl_xor(mx, 4, 16));
      mx = fmaxf(mx, __shfl_xor(mx, 8, 16));
      const float mnew = fmaxf(mrow[r], mx);
      scale[r] = __expf(mrow[r] - mnew);
      mrow[r] = mnew;
      const float p0 = __expf(s[0][r] - mnew);
      const float p1 = __expf(s[1][r] - mnew);
      s[0][r] = p0; s[1][r] = p1;
      float sum = p0 + p1;
      sum += __shfl_xor(sum, 1, 16);
      sum += __shfl_xor(sum, 2, 16);
      sum += __shfl_xor(sum, 4, 16);
      sum += __shfl_xor(sum, 8, 16);
      lrow[r] = lrow[r] * scale[r] + sum;
    }
#pragma unroll
    for (int f = 0; f < 4; f++)
#pragma unroll
      for (int r = 0; r < 8; r++) o[f][r] *= scale[r];

    // ---- C-layout -> A-layout transpose of P via per-wave LDS tile ----
#pragma unroll
    for (int r = 0; r < 8; r++) {
      psh[wave][(hi * 8 + r) * 32 + m]      = (__bf16)s[0][r];
      psh[wave][(hi * 8 + r) * 32 + 16 + m] = (__bf16)s[1][r];
    }
    v16bf pa;
    {
      const __bf16* pr = &psh[wave][m * 32];       // same-wave LDS is in-order
      v8bf c0 = *(const v8bf*)(pr + hi * 8);
      v8bf c1 = *(const v8bf*)(pr + 16 + hi * 8);
#pragma unroll
      for (int i = 0; i < 8; i++) { pa[i] = c0[i]; pa[8 + i] = c1[i]; }
    }

    // ---- O += P . V  (contraction over the 32 keys of this block) ----
#pragma unroll
    for (int f = 0; f < 4; f++) {
      // B fragment: lane holds dim n = f*16+m, contraction key = kb+hi*8+i (i<8) / kb+16+hi*8+(i-8)
      v16bf bv = load_b_frag(Vp + (size_t)(f * 16 + m) * BQ + kb + hi * 8);
      o[f] = wmma_bf16(pa, bv, o[f]);
    }
  }

  // normalize and store to [B*Q, NHD] (col = h*64 + d)
#pragma unroll
  for (int r = 0; r < 8; r++) {
    const float inv = 1.0f / lrow[r];
    const int qrow = q0 + hi * 8 + r;
    const size_t base = ((size_t)b * BQ + qrow) * NHD + h * ND;
#pragma unroll
    for (int f = 0; f < 4; f++)
      Attn[base + f * 16 + m] = (__bf16)(o[f][r] * inv);
  }
}

// ---------------------------------------------------------------------------
// Kernel 3: output projection  [BL, NHD] bf16 x Wot[N,K] bf16 -> f32 out
// ---------------------------------------------------------------------------
__global__ __launch_bounds__(128)
void out_proj_kernel(const __bf16* __restrict__ A,    // [BL, NHD] bf16
                     const __bf16* __restrict__ Wot,  // [NF(n), NHD(k)] bf16
                     float* __restrict__ Out)         // [BL, NF] f32
{
  const int lane = threadIdx.x & 31;
  const int wave = threadIdx.x >> 5;
  const int m  = lane & 15;
  const int hi = lane >> 4;
  const int row0 = blockIdx.x * 64 + wave * 16;
  const int col0 = blockIdx.y * 64;

  v8f c[4] = {};

  for (int kk = 0; kk < NHD; kk += 32) {
    const __bf16* ar = A + (size_t)(row0 + m) * NHD + kk;
    v8bf c0 = *(const v8bf*)(ar + hi * 8);
    v8bf c1 = *(const v8bf*)(ar + 16 + hi * 8);
    v16bf a;
#pragma unroll
    for (int i = 0; i < 8; i++) { a[i] = c0[i]; a[8 + i] = c1[i]; }

#pragma unroll
    for (int ns = 0; ns < 4; ns++) {
      const int n = col0 + ns * 16 + m;
      v16bf bb = load_b_frag(Wot + (size_t)n * NHD + kk + hi * 8);
      c[ns] = wmma_bf16(a, bb, c[ns]);
    }
  }

#pragma unroll
  for (int ns = 0; ns < 4; ns++)
#pragma unroll
    for (int r = 0; r < 8; r++)
      Out[(size_t)(row0 + hi * 8 + r) * NF + col0 + ns * 16 + m] = c[ns][r];
}

// ---------------------------------------------------------------------------
extern "C" void kernel_launch(void* const* d_in, const int* in_sizes, int n_in,
                              void* d_out, int out_size, void* d_ws, size_t ws_size,
                              hipStream_t stream) {
  const float* inputs_q  = (const float*)d_in[0];
  const float* inputs_kv = (const float*)d_in[1];
  const float* Wq = (const float*)d_in[2];
  const float* Wk = (const float*)d_in[3];
  const float* Wv = (const float*)d_in[4];
  const float* Wo = (const float*)d_in[5];

  const size_t elems = (size_t)NB * NH * BQ * ND;   // 4 Mi elements
  const size_t welems = (size_t)512 * 512;          // 256 Ki elements
  __bf16* qbuf = (__bf16*)d_ws;                     //  8 MB  [B][H][L][D]
  __bf16* kbuf = qbuf + elems;                      //  8 MB  [B][H][L][D]
  __bf16* vbuf = kbuf + elems;                      //  8 MB  [B][H][D][L]
  __bf16* attn = vbuf + elems;                      //  8 MB  [B*Q, NHD]
  __bf16* wqt  = attn + elems;                      // 512 KB each
  __bf16* wkt  = wqt + welems;
  __bf16* wvt  = wkt + welems;
  __bf16* wot  = wvt + welems;
  // total workspace: 34 MB

  dim3 blk(128);

  // one-shot weight transpose/convert (tiny)
  hipLaunchKernelGGL(convert_w_kernel, dim3(1024), dim3(256), 0, stream, Wq, wqt);
  hipLaunchKernelGGL(convert_w_kernel, dim3(1024), dim3(256), 0, stream, Wk, wkt);
  hipLaunchKernelGGL(convert_w_kernel, dim3(1024), dim3(256), 0, stream, Wv, wvt);
  hipLaunchKernelGGL(convert_w_kernel, dim3(1024), dim3(256), 0, stream, Wo, wot);

  dim3 gproj(BL / 64, NHD / 64);                    // (128, 8)
  hipLaunchKernelGGL(qkv_proj_kernel, gproj, blk, 0, stream, inputs_q,  wqt, qbuf, 0);
  hipLaunchKernelGGL(qkv_proj_kernel, gproj, blk, 0, stream, inputs_kv, wkt, kbuf, 0);
  hipLaunchKernelGGL(qkv_proj_kernel, gproj, blk, 0, stream, inputs_kv, wvt, vbuf, 1);

  dim3 gattn(BQ / 64, NH, NB);                      // (64, 8, 2)
  hipLaunchKernelGGL(flash_attn_kernel, gattn, blk, 0, stream, qbuf, kbuf, vbuf, attn);

  dim3 gout(BL / 64, NF / 64);                      // (128, 8)
  hipLaunchKernelGGL(out_proj_kernel, gout, blk, 0, stream, attn, wot, (float*)d_out);
}
